// SAE_36773509989203
// MI455X (gfx1250) — compile-verified, matrix operand order checked by reading, not run on hardware
//
#include <hip/hip_runtime.h>
#include <hip/hip_bf16.h>
#include <cstdint>

#define D_MODEL   768
#define NFEAT     32768
#define BATCH     1024
#define TOPK      64
#define AUXK      128
#define DEAD_AFTER 1000

typedef __attribute__((ext_vector_type(16))) __bf16 v16bf;
typedef __attribute__((ext_vector_type(8)))  float  v8f;

#if __has_builtin(__builtin_amdgcn_global_load_async_to_lds_b128) && \
    __has_builtin(__builtin_amdgcn_s_wait_asynccnt)
#define HAVE_ASYNC_LDS 1
typedef int i4v __attribute__((vector_size(16)));
#define ASYNC_B128(g, l)                                                    \
    __builtin_amdgcn_global_load_async_to_lds_b128(                         \
        (__attribute__((address_space(1))) i4v*)(g),                        \
        (__attribute__((address_space(3))) i4v*)(l), 0, 0)
#endif

static __device__ __forceinline__ unsigned f2ord(float f) {
    unsigned u = __float_as_uint(f);
    return (u & 0x80000000u) ? ~u : (u | 0x80000000u);
}

// ---------------------------------------------------------------------------
// Kernel 0: zero fvu accumulators, compute dead_condition.any()
// ---------------------------------------------------------------------------
__global__ void init_kernel(const int* __restrict__ activated_in,
                            float* __restrict__ fvu_acc, int* __restrict__ dead_any) {
    __shared__ int any;
    if (threadIdx.x == 0) { any = 0; fvu_acc[0] = 0.f; fvu_acc[1] = 0.f; }
    __syncthreads();
    int local = 0;
    for (int i = threadIdx.x; i < NFEAT; i += 256)
        local |= (activated_in[i] > DEAD_AFTER) ? 1 : 0;
    if (local) atomicOr(&any, 1);
    __syncthreads();
    if (threadIdx.x == 0) dead_any[0] = any;
}

// ---------------------------------------------------------------------------
// Kernel 1: xe = bf16( x/avg_norm*sqrt(d) - b_post + b_pre )   [1024,768]
// ---------------------------------------------------------------------------
__global__ __launch_bounds__(256) void prep_kernel(
    const float* __restrict__ x, const float* __restrict__ b_pre,
    const float* __restrict__ b_post, const float* __restrict__ avg_norm_p,
    __bf16* __restrict__ xe) {
    int i = blockIdx.x * 256 + threadIdx.x;
    if (i >= BATCH * D_MODEL) return;
    int d = i % D_MODEL;
    float avg = avg_norm_p[0];
    float v = x[i] / avg * 27.712812921102035f - b_post[d] + b_pre[d];
    xe[i] = (__bf16)v;
}

// ---------------------------------------------------------------------------
// Kernel 2: Wt[n][k] = bf16(W_enc[k][n])  -- one-shot convert + transpose
// Block: 32 n-columns x all 768 k, LDS-staged so both sides are coalesced.
// ---------------------------------------------------------------------------
__global__ __launch_bounds__(256) void wenc_prep_kernel(
    const float* __restrict__ W, __bf16* __restrict__ Wt) {
    __shared__ __bf16 t[32][776];          // row stride 1552B (97*16) -> uint4 ok
    const int tid = threadIdx.x;
    const int bn = blockIdx.x * 32;
    // load: 24 passes, each covers 32 k-rows; 8 threads x float4 per k-row
    #pragma unroll 4
    for (int p = 0; p < 24; ++p) {
        int k = p * 32 + (tid >> 3);
        int ns = (tid & 7) * 4;
        float4 f = *(const float4*)(W + (size_t)k * NFEAT + bn + ns);
        t[ns + 0][k] = (__bf16)f.x;
        t[ns + 1][k] = (__bf16)f.y;
        t[ns + 2][k] = (__bf16)f.z;
        t[ns + 3][k] = (__bf16)f.w;
    }
    __syncthreads();
    // store: 32 rows x 768 bf16 = 3072 uint4; 12 per thread, coalesced rows
    #pragma unroll 4
    for (int p = 0; p < 12; ++p) {
        int u = p * 256 + tid;
        int n = u / 48, c = (u % 48) * 16;
        uint4 d = *(const uint4*)&t[n][c];
        *(uint4*)(Wt + (size_t)(bn + n) * D_MODEL + c) = d;
    }
}

// ---------------------------------------------------------------------------
// Kernel 3: enc[1024,32768] = xe @ Wt^T  (bf16 WMMA, f32 accum)
// 128x128 block tile, BK=32, double-buffered LDS, async DMA loads when
// available. 8 waves -> each wave 64x32 (4x2 WMMA tiles).
// ---------------------------------------------------------------------------
#define BM 128
#define BN 128
#define BK 32
#define KTILES (D_MODEL / BK)   // 24
#define LDSP 40                 // bf16 row stride: 80B (multiple of 16B)

union FragBF { uint4 u[2]; v16bf v; };

__global__ __launch_bounds__(256) void encode_gemm(
    const __bf16* __restrict__ xe, const __bf16* __restrict__ Wt,
    float* __restrict__ enc) {
    __shared__ __bf16 As[2][BM][LDSP];
    __shared__ __bf16 Bs[2][BN][LDSP];

    const int tid  = threadIdx.x;
    const int lane = tid & 31;
    const int wave = tid >> 5;
    const int wm = (wave >> 2) * 64;
    const int wn = (wave & 3) * 32;
    const int bm = blockIdx.x * BM;     // M fastest-varying: 8 co-resident
    const int bn = blockIdx.y * BN;     // blocks share each W_enc tile in L2
    const int halfsel = lane >> 4;
    const int lrow = lane & 15;

    // per-thread copy coordinates: chunks of 8 bf16 (16B); 512 chunks/tile
    const int m0 = tid >> 2,          kc0 = (tid & 3) * 8;
    const int m1 = (256 + tid) >> 2,  kc1 = ((256 + tid) & 3) * 8;
    const __bf16* ga0 = xe + (size_t)(bm + m0) * D_MODEL + kc0;
    const __bf16* ga1 = xe + (size_t)(bm + m1) * D_MODEL + kc1;
    const __bf16* gb0 = Wt + (size_t)(bn + m0) * D_MODEL + kc0;
    const __bf16* gb1 = Wt + (size_t)(bn + m1) * D_MODEL + kc1;

    v8f acc[4][2];
    #pragma unroll
    for (int mi = 0; mi < 4; ++mi)
        #pragma unroll
        for (int ni = 0; ni < 2; ++ni)
            #pragma unroll
            for (int e = 0; e < 8; ++e) acc[mi][ni][e] = 0.f;

    auto issue_tile = [&](int k0, int b) {
#ifdef HAVE_ASYNC_LDS
        ASYNC_B128(ga0 + k0, &As[b][m0][kc0]);
        ASYNC_B128(ga1 + k0, &As[b][m1][kc1]);
        ASYNC_B128(gb0 + k0, &Bs[b][m0][kc0]);
        ASYNC_B128(gb1 + k0, &Bs[b][m1][kc1]);
#else
        uint4 a0 = *(const uint4*)(ga0 + k0);
        uint4 a1 = *(const uint4*)(ga1 + k0);
        uint4 b0 = *(const uint4*)(gb0 + k0);
        uint4 b1 = *(const uint4*)(gb1 + k0);
        *(uint4*)&As[b][m0][kc0] = a0;
        *(uint4*)&As[b][m1][kc1] = a1;
        *(uint4*)&Bs[b][m0][kc0] = b0;
        *(uint4*)&Bs[b][m1][kc1] = b1;
        if (k0 + BK < D_MODEL) __builtin_prefetch(gb0 + k0 + BK, 0, 1);
#endif
    };

    issue_tile(0, 0);
    for (int t = 0; t < KTILES; ++t) {
        const int buf = t & 1;
        if (t + 1 < KTILES) {
            issue_tile((t + 1) * BK, buf ^ 1);
#ifdef HAVE_ASYNC_LDS
            __builtin_amdgcn_s_wait_asynccnt(4);   // tile t complete; t+1 in flight
#endif
        } else {
#ifdef HAVE_ASYNC_LDS
            __builtin_amdgcn_s_wait_asynccnt(0);
#endif
        }
        __syncthreads();

        FragBF a[4], b[2];
        #pragma unroll
        for (int mi = 0; mi < 4; ++mi) {
            const __bf16* p = &As[buf][wm + mi * 16 + lrow][halfsel * 8];
            a[mi].u[0] = *(const uint4*)p;
            a[mi].u[1] = *(const uint4*)(p + 16);
        }
        #pragma unroll
        for (int ni = 0; ni < 2; ++ni) {
            const __bf16* p = &Bs[buf][wn + ni * 16 + lrow][halfsel * 8];
            b[ni].u[0] = *(const uint4*)p;
            b[ni].u[1] = *(const uint4*)(p + 16);
        }
        #pragma unroll
        for (int mi = 0; mi < 4; ++mi)
            #pragma unroll
            for (int ni = 0; ni < 2; ++ni)
                acc[mi][ni] = __builtin_amdgcn_wmma_f32_16x16x32_bf16(
                    false, a[mi].v, false, b[ni].v, (short)0, acc[mi][ni], false, false);
        __syncthreads();
    }

    #pragma unroll
    for (int mi = 0; mi < 4; ++mi)
        #pragma unroll
        for (int ni = 0; ni < 2; ++ni) {
            int col = bn + wn + ni * 16 + lrow;
            #pragma unroll
            for (int j = 0; j < 8; ++j) {
                int row = bm + wm + mi * 16 + j + halfsel * 8;
                enc[(size_t)row * NFEAT + col] = acc[mi][ni][j];
            }
        }
}

// ---------------------------------------------------------------------------
// Kernel 4: per-row radix-select top-64 + dead-masked top-128
// ---------------------------------------------------------------------------
#define NBINS 4096
#define CAND_CAP 2048

__global__ __launch_bounds__(256) void topk_kernel(
    const float* __restrict__ enc, const int* __restrict__ activated_in,
    float* __restrict__ tw, int* __restrict__ ti,
    float* __restrict__ aw, int* __restrict__ ai) {
    __shared__ unsigned hist[NBINS];
    __shared__ float cval[CAND_CAP];
    __shared__ int   cidx[CAND_CAP];
    __shared__ float sval[256];
    __shared__ int   spos[256];
    __shared__ int s_tb, s_cnt;

    const int tid = threadIdx.x;
    const int r = blockIdx.x;
    const float* row = enc + (size_t)r * NFEAT;
    const float NEG_INF = __uint_as_float(0xFF800000u);

    for (int phase = 0; phase < 2; ++phase) {
        const int Ksel = phase ? AUXK : TOPK;
        for (int b = tid; b < NBINS; b += 256) hist[b] = 0u;
        if (tid == 0) s_cnt = 0;
        __syncthreads();
        for (int i = tid; i < NFEAT; i += 256) {
            float v = row[i];
            if (phase && !(activated_in[i] > DEAD_AFTER)) v = NEG_INF;
            atomicAdd(&hist[f2ord(v) >> 20], 1u);
        }
        __syncthreads();
        if (tid == 0) {
            unsigned cum = 0; int tb = 0;
            for (int b = NBINS - 1; b >= 0; --b) {
                cum += hist[b];
                if (cum >= (unsigned)Ksel) { tb = b; break; }
            }
            s_tb = tb;
        }
        __syncthreads();
        const unsigned tb = (unsigned)s_tb;
        for (int i = tid; i < NFEAT; i += 256) {
            float v = row[i];
            if (phase && !(activated_in[i] > DEAD_AFTER)) v = NEG_INF;
            if ((f2ord(v) >> 20) >= tb) {
                int p = atomicAdd(&s_cnt, 1);
                if (p < CAND_CAP) { cval[p] = v; cidx[p] = i; }
            }
        }
        __syncthreads();
        const int ncand = (s_cnt < CAND_CAP) ? s_cnt : CAND_CAP;
        float* ow = phase ? (aw + (size_t)r * AUXK) : (tw + (size_t)r * TOPK);
        int*   oi = phase ? (ai + (size_t)r * AUXK) : (ti + (size_t)r * TOPK);
        for (int j = 0; j < Ksel; ++j) {
            float bv = NEG_INF; int bp = -1;
            for (int i = tid; i < ncand; i += 256)
                if (bp < 0 || cval[i] > bv) { bv = cval[i]; bp = i; }
            sval[tid] = bv; spos[tid] = bp;
            __syncthreads();
            for (int s = 128; s > 0; s >>= 1) {
                if (tid < s) {
                    if (spos[tid] < 0 || (spos[tid + s] >= 0 && sval[tid + s] > sval[tid])) {
                        sval[tid] = sval[tid + s]; spos[tid] = spos[tid + s];
                    }
                }
                __syncthreads();
            }
            if (tid == 0) {
                int w = spos[0];
                float v = (w >= 0) ? sval[0] : NEG_INF;
                ow[j] = (w >= 0) ? ((phase && v == NEG_INF) ? 0.0f : v) : 0.0f;
                oi[j] = (w >= 0) ? cidx[w] : 0;
                if (w >= 0) cval[w] = NEG_INF;
            }
            __syncthreads();
        }
        __syncthreads();
    }
}

// ---------------------------------------------------------------------------
// Kernel 5: sparse decode (main + aux), y/loss/fvu partials
// ---------------------------------------------------------------------------
__global__ __launch_bounds__(256) void decode_kernel(
    const float* __restrict__ x, const float* __restrict__ W_dec,
    const float* __restrict__ b_post, const float* __restrict__ avg_norm_p,
    const float* __restrict__ tw, const int* __restrict__ ti,
    const float* __restrict__ aw, const int* __restrict__ ai,
    const int* __restrict__ dead_any,
    float* __restrict__ y_out, float* __restrict__ loss_out,
    float* __restrict__ fvu_acc) {
    const int tid = threadIdx.x, r = blockIdx.x;
    float acc[3] = {0.f, 0.f, 0.f}, aacc[3] = {0.f, 0.f, 0.f};
    const int d0 = tid, d1 = tid + 256, d2 = tid + 512;

    for (int k = 0; k < TOPK; ++k) {
        float w = tw[r * TOPK + k];
        const float* wd = W_dec + (size_t)ti[r * TOPK + k] * D_MODEL;
        acc[0] += w * wd[d0]; acc[1] += w * wd[d1]; acc[2] += w * wd[d2];
    }
    for (int k = 0; k < AUXK; ++k) {
        float w = aw[r * AUXK + k];
        const float* wd = W_dec + (size_t)ai[r * AUXK + k] * D_MODEL;
        aacc[0] += w * wd[d0]; aacc[1] += w * wd[d1]; aacc[2] += w * wd[d2];
    }
    const float avg = avg_norm_p[0];
    const float tgt = 27.712812921102035f;
    float se = 0.f, sae = 0.f, sxx = 0.f;
    #pragma unroll
    for (int j = 0; j < 3; ++j) {
        int d = tid + j * 256;
        float xn = x[(size_t)r * D_MODEL + d] / avg * tgt;
        float yn = acc[j] + b_post[d];
        y_out[(size_t)r * D_MODEL + d] = yn / tgt * avg;
        float e = xn - yn; se += e * e; sxx += xn * xn;
        float ae = yn - xn - aacc[j]; sae += ae * ae;
    }
    __shared__ float r1[256], r2[256], r3[256];
    r1[tid] = se; r2[tid] = sae; r3[tid] = sxx;
    __syncthreads();
    for (int s = 128; s > 0; s >>= 1) {
        if (tid < s) { r1[tid] += r1[tid + s]; r2[tid] += r2[tid + s]; r3[tid] += r3[tid + s]; }
        __syncthreads();
    }
    if (tid == 0) {
        float recon = r1[0] * (1.0f / (float)D_MODEL);
        float auxl  = dead_any[0] ? r2[0] * (1.0f / (float)D_MODEL) : 0.f;
        loss_out[r] = recon + 0.03125f * auxl;
        atomicAdd(&fvu_acc[0], r1[0]);
        atomicAdd(&fvu_acc[1], r3[0]);
    }
}

__global__ void fvu_kernel(const float* __restrict__ fvu_acc, float* __restrict__ out) {
    out[0] = fvu_acc[0] / fvu_acc[1];
}

// ---------------------------------------------------------------------------
extern "C" void kernel_launch(void* const* d_in, const int* in_sizes, int n_in,
                              void* d_out, int out_size, void* d_ws, size_t ws_size,
                              hipStream_t stream) {
    const float* x            = (const float*)d_in[0];
    const float* W_enc        = (const float*)d_in[1];
    const float* W_dec        = (const float*)d_in[2];
    const float* b_pre        = (const float*)d_in[3];
    const float* b_post       = (const float*)d_in[4];
    const float* avg_norm     = (const float*)d_in[5];
    const int*   activated_in = (const int*)d_in[6];

    char* ws = (char*)d_ws;
    size_t off = 0;
    float*  enc = (float*)(ws + off);  off += (size_t)BATCH * NFEAT * 4;      // 128 MB
    __bf16* Wt  = (__bf16*)(ws + off); off += (size_t)NFEAT * D_MODEL * 2;    // 48 MB
    __bf16* xe  = (__bf16*)(ws + off); off += (size_t)BATCH * D_MODEL * 2;    // 1.5 MB
    float*  tw  = (float*)(ws + off);  off += (size_t)BATCH * TOPK * 4;
    int*    ti  = (int*)(ws + off);    off += (size_t)BATCH * TOPK * 4;
    float*  aw  = (float*)(ws + off);  off += (size_t)BATCH * AUXK * 4;
    int*    ai  = (int*)(ws + off);    off += (size_t)BATCH * AUXK * 4;
    float*  fvu_acc  = (float*)(ws + off); off += 8;
    int*    dead_any = (int*)(ws + off);   off += 4;

    float* y    = (float*)d_out;
    float* loss = y + (size_t)BATCH * D_MODEL;
    float* fvu  = loss + BATCH;

    init_kernel<<<1, 256, 0, stream>>>(activated_in, fvu_acc, dead_any);
    prep_kernel<<<(BATCH * D_MODEL + 255) / 256, 256, 0, stream>>>(x, b_pre, b_post, avg_norm, xe);
    wenc_prep_kernel<<<NFEAT / 32, 256, 0, stream>>>(W_enc, Wt);
    dim3 gemm_grid(BATCH / BM, NFEAT / BN);   // M fastest -> W_enc tile reuse in L2
    encode_gemm<<<gemm_grid, 256, 0, stream>>>(xe, Wt, enc);
    topk_kernel<<<BATCH, 256, 0, stream>>>(enc, activated_in, tw, ti, aw, ai);
    decode_kernel<<<BATCH, 256, 0, stream>>>(x, W_dec, b_post, avg_norm,
                                             tw, ti, aw, ai, dead_any, y, loss, fvu_acc);
    fvu_kernel<<<1, 1, 0, stream>>>(fvu_acc, fvu);
}
